// Attention_41996190220943
// MI455X (gfx1250) — compile-verified, matrix operand order checked by reading, not run on hardware
//
#include <hip/hip_runtime.h>
#include <hip/hip_bf16.h>
#include <cmath>

// ---------------------------------------------------------------------------
// GQA attention forward for MI455X (gfx1250): BF16 WMMA everywhere,
// flash-style fused attention, LDS-staged GEMMs, async global->LDS staging
// (GLOBAL_LOAD_ASYNC_TO_LDS_B128, tracked by ASYNCcnt) where available.
// ---------------------------------------------------------------------------

typedef __bf16 bf16;
typedef bf16  v16bf __attribute__((ext_vector_type(16)));
typedef float v8f   __attribute__((ext_vector_type(8)));
typedef int   v4i   __attribute__((ext_vector_type(4)));

constexpr int B_ = 2, S_ = 2048, D_ = 2048, H_ = 16, G_ = 4, DH_ = 128;
constexpr float INV_SQRT_DH = 0.08838834764831845f;   // 1/sqrt(128)
constexpr float LOG_BASE    = 13.815510557964274f;    // ln(1e6)

__device__ __forceinline__ v8f v8f_zero() {
  v8f z = {0.f, 0.f, 0.f, 0.f, 0.f, 0.f, 0.f, 0.f};
  return z;
}

__device__ __forceinline__ v8f wmma_bf16(v16bf a, v16bf b, v8f c) {
  return __builtin_amdgcn_wmma_f32_16x16x32_bf16(
      /*neg_a=*/false, a, /*neg_b=*/false, b,
      /*c_mod=*/(short)0, c, /*reuse_a=*/false, /*reuse_b=*/false);
}

// ---- global -> LDS 16-byte copy: async path (ASYNCcnt) or VGPR fallback ----
#if defined(__has_builtin)
#if __has_builtin(__builtin_amdgcn_global_load_async_to_lds_b128)
#define ASYNC_LDS 1
#endif
#endif
#ifndef ASYNC_LDS
#define ASYNC_LDS 0
#endif

typedef __attribute__((address_space(1))) v4i as1_v4i;  // global
typedef __attribute__((address_space(3))) v4i as3_v4i;  // LDS

__device__ __forceinline__ void cp16_g2l(bf16* lds_dst, const bf16* gsrc) {
#if ASYNC_LDS
  __builtin_amdgcn_global_load_async_to_lds_b128(
      (as1_v4i*)gsrc, (as3_v4i*)lds_dst, 0, 0);
#else
  *(uint4*)lds_dst = *(const uint4*)gsrc;
#endif
}

__device__ __forceinline__ void cp16_wait() {
#if ASYNC_LDS
#if __has_builtin(__builtin_amdgcn_s_wait_asynccnt)
  __builtin_amdgcn_s_wait_asynccnt(0);
#else
  asm volatile("s_wait_asynccnt 0x0" ::: "memory");
#endif
#endif
}

// Load a 16-bit WMMA operand fragment from an [rows][K] (K-contiguous) tile.
// A operand: rows = M;  B operand: rows = N (i.e. B^T storage).
__device__ __forceinline__ v16bf load_frag_nk(const bf16* base, int row,
                                              int stride, int k0) {
  int lane  = threadIdx.x & 31;
  int r     = lane & 15;
  int khalf = (lane >> 4) & 1;
  const bf16* p = base + (size_t)(row + r) * stride + k0;
  v16bf f;
#pragma unroll
  for (int v = 0; v < 8; ++v) {
    int k = ((v & 4) ? 16 : 0) + khalf * 8 + (v & 3) * 2;
    f[2 * v]     = p[k];
    f[2 * v + 1] = p[k + 1];
  }
  return f;
}

// ---------------------------------------------------------------------------
__global__ void f32_to_bf16_kernel(const float* __restrict__ in,
                                   bf16* __restrict__ out, int n) {
  int i = blockIdx.x * blockDim.x + threadIdx.x;
  if (i < n) out[i] = (bf16)in[i];
}

// RoPE in place on a [rows, heads*128] bf16 buffer; rows = B*S (s = row % S).
__global__ void rope_bf16_kernel(bf16* __restrict__ buf, int rows, int heads) {
  int idx   = blockIdx.x * blockDim.x + threadIdx.x;
  int ppr   = heads * (DH_ / 2);
  int total = rows * ppr;
  if (idx >= total) return;
  int row = idx / ppr;
  int pr  = idx - row * ppr;
  int i   = pr & (DH_ / 2 - 1);
  int hh  = pr / (DH_ / 2);
  int s   = row & (S_ - 1);
  float omega = __expf(-(2.0f * (float)i / (float)DH_) * LOG_BASE);
  float ang = (float)s * omega;
  float sn, cs;
  __sincosf(ang, &sn, &cs);
  size_t off = (size_t)row * heads * DH_ + (size_t)hh * DH_ + 2 * i;
  float xe = (float)buf[off];
  float xo = (float)buf[off + 1];
  buf[off]     = (bf16)(xe * cs - xo * sn);
  buf[off + 1] = (bf16)(xo * cs + xe * sn);
}

// ---------------------------------------------------------------------------
// C[M,N] = A[M,K] @ W[N,K]^T + bias, bf16 in, fp32 accumulate.
// 4 waves / block; block computes a 64x64 tile; A and B k-slabs staged in LDS
// with coalesced 16B transfers; B slab reused by all 4 waves.
template <typename OutT>
__global__ __launch_bounds__(128)
void gemm_nt_bias(const bf16* __restrict__ A, const bf16* __restrict__ W,
                  const float* __restrict__ bias, OutT* __restrict__ C,
                  int M, int N, int K) {
  __shared__ alignas(16) bf16 aTile[64 * 32];
  __shared__ alignas(16) bf16 bTile[64 * 32];

  int tid  = threadIdx.x;
  int wave = tid >> 5;
  int lane = tid & 31;
  int row0 = blockIdx.x * 64;
  int col0 = blockIdx.y * 64;

  v8f acc[4];
#pragma unroll
  for (int nb = 0; nb < 4; ++nb) acc[nb] = v8f_zero();

  for (int k0 = 0; k0 < K; k0 += 32) {
    // stage 64x32 A and 64x32 B slabs (256 x 16B chunks each, 2 per thread)
#pragma unroll
    for (int c = tid; c < 256; c += 128) {
      int r = c >> 2;           // tile row
      int e = (c & 3) * 8;      // element offset in row (16B chunks)
      cp16_g2l(aTile + r * 32 + e, A + (size_t)(row0 + r) * K + k0 + e);
      cp16_g2l(bTile + r * 32 + e, W + (size_t)(col0 + r) * K + k0 + e);
    }
    cp16_wait();
    __syncthreads();

    v16bf a = load_frag_nk(aTile, wave * 16, 32, 0);
#pragma unroll
    for (int nb = 0; nb < 4; ++nb) {
      v16bf b = load_frag_nk(bTile, nb * 16, 32, 0);
      acc[nb] = wmma_bf16(a, b, acc[nb]);
    }
    __syncthreads();
  }

  int nloc  = lane & 15;
  int mhalf = lane >> 4;
  int rowW  = row0 + wave * 16;
#pragma unroll
  for (int nb = 0; nb < 4; ++nb) {
    int n = col0 + nb * 16 + nloc;
    float bv = bias[n];
#pragma unroll
    for (int r = 0; r < 8; ++r) {
      int m = rowW + mhalf * 8 + r;
      C[(size_t)m * N + n] = (OutT)(acc[nb][r] + bv);
    }
  }
}

// ---------------------------------------------------------------------------
// Flash attention: one wave per (b, h, 16-row query tile).
// Q: [B,S,H,DH] bf16 (RoPE'd)   K,V: [B,S,G,DH] bf16 (K RoPE'd)
// O: [B,S,H,DH] bf16
__global__ __launch_bounds__(32)
void flash_attn_kernel(const bf16* __restrict__ Q, const bf16* __restrict__ Kc,
                       const bf16* __restrict__ Vc, bf16* __restrict__ O) {
  int mt = blockIdx.x;
  int bh = blockIdx.y;
  int b  = bh / H_;
  int h  = bh - b * H_;
  int g  = h / (H_ / G_);
  int lane = threadIdx.x;

  __shared__ alignas(16) bf16 qs[16 * DH_];    // [qrow][dh]
  __shared__ alignas(16) bf16 ks[32 * DH_];    // [key ][dh]
  __shared__ alignas(16) bf16 vts[DH_ * 32];   // [dh  ][key]
  __shared__ alignas(16) bf16 ps[16 * 32];     // [qrow][key]

  // stage Q tile: 16 rows x 128 dh = 256 x 16B chunks
#pragma unroll
  for (int c = lane; c < 256; c += 32) {
    int r = c >> 4, e = (c & 15) * 8;
    cp16_g2l(qs + r * DH_ + e,
             Q + (size_t)(b * S_ + mt * 16 + r) * D_ + h * DH_ + e);
  }
  cp16_wait();
  __syncthreads();

  v8f acc[8];
#pragma unroll
  for (int nb = 0; nb < 8; ++nb) acc[nb] = v8f_zero();
  float m_run[8], l_run[8];
#pragma unroll
  for (int r = 0; r < 8; ++r) { m_run[r] = -__builtin_inff(); l_run[r] = 0.f; }

  int nloc  = lane & 15;
  int mhalf = lane >> 4;
  int qrow0 = mt * 16 + mhalf * 8;

  int kt_end = (mt * 16 + 15) / 32;
  for (int kt = 0; kt <= kt_end; ++kt) {
    // K tile: 32 rows x 128 dh = 512 x 16B chunks, natural layout
#pragma unroll
    for (int c = lane; c < 512; c += 32) {
      int kr = c >> 4, e = (c & 15) * 8;
      cp16_g2l(ks + kr * DH_ + e,
               Kc + (size_t)(b * S_ + kt * 32 + kr) * (G_ * DH_) + g * DH_ + e);
    }
    // V tile transposed: wide global read, 2-byte LDS scatter
#pragma unroll
    for (int c = lane; c < 512; c += 32) {
      int kr = c >> 4, e = (c & 15) * 8;
      uint4 t = *(const uint4*)(Vc + (size_t)(b * S_ + kt * 32 + kr) * (G_ * DH_)
                                + g * DH_ + e);
      const bf16* tb = reinterpret_cast<const bf16*>(&t);
#pragma unroll
      for (int j = 0; j < 8; ++j) vts[(e + j) * 32 + kr] = tb[j];
    }
    cp16_wait();
    __syncthreads();

    // scores: 16x32 = two 16x16 WMMA blocks, K = DH = 128 in 4 steps
    v8f s0 = v8f_zero(), s1 = v8f_zero();
#pragma unroll
    for (int kk = 0; kk < 4; ++kk) {
      v16bf a  = load_frag_nk(qs, 0, DH_, kk * 32);
      v16bf b0 = load_frag_nk(ks, 0, DH_, kk * 32);
      s0 = wmma_bf16(a, b0, s0);
      v16bf b1 = load_frag_nk(ks, 16, DH_, kk * 32);
      s1 = wmma_bf16(a, b1, s1);
    }

    // online softmax over this tile's 32 keys
    float alpha[8];
#pragma unroll
    for (int r = 0; r < 8; ++r) {
      int qi  = qrow0 + r;
      int k0i = kt * 32 + nloc;
      int k1i = kt * 32 + 16 + nloc;
      float v0 = (k0i > qi) ? -__builtin_inff() : s0[r] * INV_SQRT_DH;
      float v1 = (k1i > qi) ? -__builtin_inff() : s1[r] * INV_SQRT_DH;
      float mx = fmaxf(v0, v1);
#pragma unroll
      for (int off = 1; off < 16; off <<= 1)
        mx = fmaxf(mx, __shfl_xor(mx, off, 32));
      float mn = fmaxf(m_run[r], mx);
      alpha[r] = __expf(m_run[r] - mn);
      float p0 = __expf(v0 - mn);
      float p1 = __expf(v1 - mn);
      float rs = p0 + p1;
#pragma unroll
      for (int off = 1; off < 16; off <<= 1)
        rs += __shfl_xor(rs, off, 32);
      l_run[r] = l_run[r] * alpha[r] + rs;
      m_run[r] = mn;
      int prow = mhalf * 8 + r;
      ps[prow * 32 + nloc]      = (bf16)p0;
      ps[prow * 32 + 16 + nloc] = (bf16)p1;
    }
#pragma unroll
    for (int nb = 0; nb < 8; ++nb)
#pragma unroll
      for (int r = 0; r < 8; ++r) acc[nb][r] *= alpha[r];
    __syncthreads();

    // out += P(16x32) @ V(32x128)
    v16bf pA = load_frag_nk(ps, 0, 32, 0);
#pragma unroll
    for (int nb = 0; nb < 8; ++nb) {
      v16bf vB = load_frag_nk(vts, nb * 16, 32, 0);
      acc[nb] = wmma_bf16(pA, vB, acc[nb]);
    }
    __syncthreads();
  }

  // epilogue: normalize, store [B,S,H,DH]
#pragma unroll
  for (int nb = 0; nb < 8; ++nb) {
#pragma unroll
    for (int r = 0; r < 8; ++r) {
      int m = qrow0 + r;
      float o = acc[nb][r] / l_run[r];
      O[(size_t)(b * S_ + m) * D_ + h * DH_ + nb * 16 + nloc] = (bf16)o;
    }
  }
}

// ---------------------------------------------------------------------------
extern "C" void kernel_launch(void* const* d_in, const int* in_sizes, int n_in,
                              void* d_out, int out_size, void* d_ws,
                              size_t ws_size, hipStream_t stream) {
  (void)in_sizes; (void)n_in; (void)out_size; (void)ws_size;
  const float* x  = (const float*)d_in[0];
  const float* wq = (const float*)d_in[1];
  const float* bq = (const float*)d_in[2];
  const float* wk = (const float*)d_in[3];
  const float* bk = (const float*)d_in[4];
  const float* wv = (const float*)d_in[5];
  const float* bv = (const float*)d_in[6];
  const float* wo = (const float*)d_in[7];
  const float* bo = (const float*)d_in[8];
  // d_in[9] = causal mask: recomputed analytically in-kernel.

  const int M  = B_ * S_;          // 4096
  const int NQ = D_;               // 2048
  const int NK = G_ * DH_;         // 512

  char* ws = (char*)d_ws;
  size_t off = 0;
  auto alloc = [&](size_t elems) {
    bf16* p = (bf16*)(ws + off);
    off += elems * sizeof(bf16);
    return p;
  };
  bf16* xbf = alloc((size_t)M * D_);
  bf16* wqb = alloc((size_t)NQ * D_);
  bf16* wkb = alloc((size_t)NK * D_);
  bf16* wvb = alloc((size_t)NK * D_);
  bf16* wob = alloc((size_t)D_ * D_);
  bf16* qb  = alloc((size_t)M * NQ);
  bf16* kb  = alloc((size_t)M * NK);
  bf16* vb  = alloc((size_t)M * NK);
  bf16* ab  = alloc((size_t)M * D_);

  auto cvt = [&](const float* src, bf16* dst, int n) {
    f32_to_bf16_kernel<<<(n + 255) / 256, 256, 0, stream>>>(src, dst, n);
  };
  cvt(x,  xbf, M * D_);
  cvt(wq, wqb, NQ * D_);
  cvt(wk, wkb, NK * D_);
  cvt(wv, wvb, NK * D_);
  cvt(wo, wob, D_ * D_);

  dim3 blk(128);
  gemm_nt_bias<bf16><<<dim3(M / 64, NQ / 64), blk, 0, stream>>>(
      xbf, wqb, bq, qb, M, NQ, D_);
  gemm_nt_bias<bf16><<<dim3(M / 64, NK / 64), blk, 0, stream>>>(
      xbf, wkb, bk, kb, M, NK, D_);
  gemm_nt_bias<bf16><<<dim3(M / 64, NK / 64), blk, 0, stream>>>(
      xbf, wvb, bv, vb, M, NK, D_);

  int qpairs = M * H_ * (DH_ / 2);
  rope_bf16_kernel<<<(qpairs + 255) / 256, 256, 0, stream>>>(qb, M, H_);
  int kpairs = M * G_ * (DH_ / 2);
  rope_bf16_kernel<<<(kpairs + 255) / 256, 256, 0, stream>>>(kb, M, G_);

  flash_attn_kernel<<<dim3(S_ / 16, B_ * H_), 32, 0, stream>>>(qb, kb, vb, ab);

  gemm_nt_bias<float><<<dim3(M / 64, D_ / 64), blk, 0, stream>>>(
      ab, wob, bo, (float*)d_out, M, D_, D_);
}
